// QLSTMGen_65481071403245
// MI455X (gfx1250) — compile-verified, matrix operand order checked by reading, not run on hardware
//
#include <hip/hip_runtime.h>
#include <hip/hip_bf16.h>
#include <math.h>

typedef __bf16 bf16_t;
typedef bf16_t v16bf __attribute__((ext_vector_type(16)));
typedef float  v8f   __attribute__((ext_vector_type(8)));

#define T_STEPS 256
#define BATCH   128
#define IDIM    256
#define HDIM    256
#define NQD     8
#define NGCOL   32   // 4 gates * NQ

// K index inside a 16-bit 16x32 A/B fragment for a given lane and half-slot.
// ISA 7.12.2: lanes 0-15 hold K=0..7 (v0..3) and 16..23 (v4..7); lanes 16-31
// hold K=8..15 and 24..31. Two 16-bit halves per VGPR.
__device__ __forceinline__ int frag_k(int lane, int h) {
  int v = h >> 1, s = h & 1;
  int k = (v < 4) ? (v * 2 + s) : (16 + (v - 4) * 2 + s);
  return k + ((lane & 16) ? 8 : 0);
}

// Load a 16x32 (MxK) fragment from a row-major bf16 matrix. Also used for B
// when the source is W[n][k] (i.e. B^T row-major): lane&15 selects the N col.
__device__ __forceinline__ v16bf load_frag_bf16(const bf16_t* base, int row0,
                                                int stride, int k0, int lane) {
  v16bf f;
  int r = row0 + (lane & 15);
#pragma unroll
  for (int h = 0; h < 16; ++h)
    f[h] = base[r * stride + k0 + frag_k(lane, h)];
  return f;
}

// ---------------------------------------------------------------------------
// Kernel 1: px[t*B+b, n] = x[t,b,:] @ Wx_g[q,:].T + b_g[q]   (n = g*8+q)
// Fully parallel over T*B = 32768 rows. 512 blocks x 256 threads (8 waves);
// each wave produces one 16x16 C tile with 8 K-step bf16 WMMAs.
// ---------------------------------------------------------------------------
__global__ __launch_bounds__(256) void qlstm_px_kernel(
    const float* __restrict__ x,
    const float* __restrict__ Wf, const float* __restrict__ bfv,
    const float* __restrict__ Wi, const float* __restrict__ biv,
    const float* __restrict__ Wu, const float* __restrict__ buv,
    const float* __restrict__ Wo, const float* __restrict__ bov,
    float* __restrict__ px) {
  __shared__ bf16_t sW[NGCOL * IDIM];
  __shared__ float  sBias[NGCOL];
  const int tid  = threadIdx.x;
  const int lane = tid & 31;
  const int wv   = tid >> 5;  // 0..7

  for (int idx = tid; idx < NGCOL * IDIM; idx += 256) {
    int n = idx >> 8, k = idx & 255;
    int g = n >> 3, q = n & 7;
    const float* Wg = (g == 0) ? Wf : (g == 1) ? Wi : (g == 2) ? Wu : Wo;
    sW[idx] = (bf16_t)Wg[q * (IDIM + HDIM) + k];
  }
  if (tid < NGCOL) {
    int g = tid >> 3, q = tid & 7;
    const float* bg = (g == 0) ? bfv : (g == 1) ? biv : (g == 2) ? buv : bov;
    sBias[tid] = bg[q];
  }
  __syncthreads();

  const int mtile = wv & 3;       // 4 row tiles of 16 -> 64 rows per block
  const int nt    = wv >> 2;      // 2 col tiles of 16 -> 32 cols
  const int rbase = blockIdx.x * 64 + mtile * 16;

  v8f acc = {};
#pragma unroll
  for (int kt = 0; kt < IDIM / 32; ++kt) {
    v16bf a;
    {
      int r = rbase + (lane & 15);
#pragma unroll
      for (int h = 0; h < 16; ++h)
        a[h] = (bf16_t)x[r * IDIM + kt * 32 + frag_k(lane, h)];
    }
    v16bf b = load_frag_bf16(sW, nt * 16, IDIM, kt * 32, lane);
    acc = __builtin_amdgcn_wmma_f32_16x16x32_bf16(false, a, false, b,
                                                  (short)0, acc, false, false);
  }
#pragma unroll
  for (int r = 0; r < 8; ++r) {
    int mrow = rbase + r + ((lane & 16) ? 8 : 0);
    int ncol = nt * 16 + (lane & 15);
    px[mrow * NGCOL + ncol] = acc[r] + sBias[ncol];
  }
}

// ---------------------------------------------------------------------------
// Kernel 2: persistent recurrence on ONE WGP. 512 threads = 16 waves.
// All recurrent state lives in LDS (~296 KB of the WGP's 320 KB):
//   sWh  bf16 [32][256]   recurrent weights (B^T layout)
//   sHx  bf16 [128][256]  hidden state (WMMA A source)
//   sAct bf16 [128][32]   post-activation quantum expectations
//   sWqF bf16 [16][4][32][16] prebuilt zero-padded Wq B-fragments (K=32)
//   sCx  f32  [128][256]  cell state
//   sZ   f32  [128][32]   gate pre-activations
// ---------------------------------------------------------------------------
__global__ __launch_bounds__(512) void qlstm_rec_kernel(
    const float* __restrict__ px,
    const float* __restrict__ Wf, const float* __restrict__ Wi,
    const float* __restrict__ Wu, const float* __restrict__ Wo,
    const float* __restrict__ Wq, const float* __restrict__ bq,
    float* __restrict__ outs, float* __restrict__ hx_out,
    float* __restrict__ cx_out) {
  extern __shared__ __align__(16) char smem_raw[];
  bf16_t* sWh  = (bf16_t*)smem_raw;                 // 32*256
  bf16_t* sHx  = sWh + NGCOL * HDIM;                // 128*256
  bf16_t* sAct = sHx + BATCH * HDIM;                // 128*32
  bf16_t* sWqF = sAct + BATCH * NGCOL;              // 16*4*32*16
  float*  sCx  = (float*)(sWqF + 16 * 4 * 32 * 16); // 128*256
  float*  sZ   = sCx + BATCH * HDIM;                // 128*32

  const int tid  = threadIdx.x;
  const int lane = tid & 31;
  const int wv   = tid >> 5;  // 0..15

  // ---- init: zero state, pack recurrent weights, build Wq B-fragments ----
  for (int i = tid; i < BATCH * HDIM; i += 512) {
    sHx[i] = (bf16_t)0.f;
    sCx[i] = 0.f;
  }
  for (int idx = tid; idx < NGCOL * HDIM; idx += 512) {
    int n = idx >> 8, k = idx & 255;
    int g = n >> 3, q = n & 7;
    const float* Wg = (g == 0) ? Wf : (g == 1) ? Wi : (g == 2) ? Wu : Wo;
    sWh[idx] = (bf16_t)Wg[q * (IDIM + HDIM) + IDIM + k];
  }
  // B frag for h-tile `ntile`, gate g: nonzero only for K in [8g, 8g+8),
  // value Wq[h, K-8g]. Stored per-lane-contiguous for fast per-step reload.
  for (int idx = tid; idx < 16 * 4 * 32 * 16; idx += 512) {
    int h     = idx & 15;
    int ln    = (idx >> 4) & 31;
    int g     = (idx >> 9) & 3;
    int ntile = idx >> 11;
    int k     = frag_k(ln, h);
    int hcol  = ntile * 16 + (ln & 15);
    float v = 0.f;
    if (k >= g * 8 && k < g * 8 + 8) v = Wq[hcol * NQD + (k - g * 8)];
    sWqF[idx] = (bf16_t)v;
  }
  const float bq_l = bq[wv * 16 + (lane & 15)];
  __syncthreads();

  for (int t = 0; t < T_STEPS; ++t) {
    // ---- Phase A: Z = hx @ Wh^T + px[t]  ([128,256]x[256,32]) ----
    {
      const int mtile = wv & 7;
      const int nt    = wv >> 3;
      v8f acc = {};
#pragma unroll
      for (int kt = 0; kt < HDIM / 32; ++kt) {
        v16bf a = load_frag_bf16(sHx, mtile * 16, HDIM, kt * 32, lane);
        v16bf b = load_frag_bf16(sWh, nt * 16, HDIM, kt * 32, lane);
        acc = __builtin_amdgcn_wmma_f32_16x16x32_bf16(
            false, a, false, b, (short)0, acc, false, false);
      }
#pragma unroll
      for (int r = 0; r < 8; ++r) {
        int mrow = mtile * 16 + r + ((lane & 16) ? 8 : 0);
        int ncol = nt * 16 + (lane & 15);
        sZ[mrow * NGCOL + ncol] =
            acc[r] + px[(t * BATCH + mrow) * NGCOL + ncol];
      }
      if (t + 1 < T_STEPS)  // warm L2/WGP$ for next step's px slice
        __builtin_prefetch(&px[((t + 1) * BATCH) * NGCOL], 0, 1);
    }
    __syncthreads();

    // ---- Phase B: expect = -cumprod(cos(z)); sigmoid (f,i,o) / tanh (u) ----
    {
      int b = tid >> 2, g = tid & 3;
      float c = 1.f;
#pragma unroll
      for (int q = 0; q < NQD; ++q) {
        float theta = sZ[b * NGCOL + g * 8 + q];
        c *= cosf(theta);
        float e = -c;
        float a = (g == 2) ? tanhf(e) : (1.f / (1.f + expf(-e)));
        sAct[b * NGCOL + g * 8 + q] = (bf16_t)a;
      }
    }
    __syncthreads();

    // ---- Phase C: gate projections (K=32, padded Wq) + LSTM cell update ----
    {
      v16bf Bf[4];
#pragma unroll
      for (int g = 0; g < 4; ++g) {
        const bf16_t* src = sWqF + ((wv * 4 + g) * 32 + lane) * 16;
#pragma unroll
        for (int h = 0; h < 16; ++h) Bf[g][h] = src[h];
      }
      const int hcol = wv * 16 + (lane & 15);
#pragma unroll
      for (int m = 0; m < 8; ++m) {
        v16bf a = load_frag_bf16(sAct, m * 16, NGCOL, 0, lane);
        v8f cf = {}, ci = {}, cu = {}, co = {};
        cf = __builtin_amdgcn_wmma_f32_16x16x32_bf16(false, a, false, Bf[0],
                                                     (short)0, cf, false, false);
        ci = __builtin_amdgcn_wmma_f32_16x16x32_bf16(false, a, false, Bf[1],
                                                     (short)0, ci, false, false);
        cu = __builtin_amdgcn_wmma_f32_16x16x32_bf16(false, a, false, Bf[2],
                                                     (short)0, cu, false, false);
        co = __builtin_amdgcn_wmma_f32_16x16x32_bf16(false, a, false, Bf[3],
                                                     (short)0, co, false, false);
#pragma unroll
        for (int r = 0; r < 8; ++r) {
          int brow = m * 16 + r + ((lane & 16) ? 8 : 0);
          float fg = cf[r] + bq_l;
          float ig = ci[r] + bq_l;
          float ug = cu[r] + bq_l;
          float og = co[r] + bq_l;
          float cnew = fg * sCx[brow * HDIM + hcol] + ig * ug;
          sCx[brow * HDIM + hcol] = cnew;
          float hnew = og * tanhf(cnew);
          sHx[brow * HDIM + hcol] = (bf16_t)hnew;
          outs[((size_t)t * BATCH + brow) * HDIM + hcol] = hnew;
          if (t == T_STEPS - 1) hx_out[brow * HDIM + hcol] = hnew;
        }
      }
    }
    __syncthreads();
  }

  for (int i = tid; i < BATCH * HDIM; i += 512) cx_out[i] = sCx[i];
}

// ---------------------------------------------------------------------------
extern "C" void kernel_launch(void* const* d_in, const int* in_sizes, int n_in,
                              void* d_out, int out_size, void* d_ws,
                              size_t ws_size, hipStream_t stream) {
  const float* x   = (const float*)d_in[0];
  const float* Wf  = (const float*)d_in[1];
  const float* bfv = (const float*)d_in[2];
  const float* Wi  = (const float*)d_in[3];
  const float* biv = (const float*)d_in[4];
  const float* Wu  = (const float*)d_in[5];
  const float* buv = (const float*)d_in[6];
  const float* Wo  = (const float*)d_in[7];
  const float* bov = (const float*)d_in[8];
  const float* Wq  = (const float*)d_in[9];
  const float* bq  = (const float*)d_in[10];

  float* outs   = (float*)d_out;
  float* hx_out = outs + (size_t)T_STEPS * BATCH * HDIM;
  float* cx_out = hx_out + (size_t)BATCH * HDIM;
  float* px     = (float*)d_ws;  // [T*B, 32] f32 = 4 MB

  qlstm_px_kernel<<<(T_STEPS * BATCH) / 64, 256, 0, stream>>>(
      x, Wf, bfv, Wi, biv, Wu, buv, Wo, bov, px);

  size_t smem =
      (size_t)(NGCOL * HDIM + BATCH * HDIM + BATCH * NGCOL + 16 * 4 * 32 * 16) *
          sizeof(bf16_t) +
      (size_t)(BATCH * HDIM + BATCH * NGCOL) * sizeof(float);  // ~296 KB
  qlstm_rec_kernel<<<1, 512, smem, stream>>>(px, Wf, Wi, Wu, Wo, Wq, bq, outs,
                                             hx_out, cx_out);
}
